// Encoder_68367289417899
// MI455X (gfx1250) — matrix operand order repaired and checked by reading, not compiled, not run on hardware
//
#include <hip/hip_runtime.h>
#include <hip/hip_bf16.h>

#if defined(__gfx1250__) && __has_builtin(__builtin_amdgcn_global_load_async_to_lds_b32)
#define ASYNC_STAGE 1
#endif

// ---------------------------------------------------------------------------
// CDNA5 WMMA types (wave32): v_wmma_f32_16x16x32_bf16
// ---------------------------------------------------------------------------
typedef __bf16 bf16_t;
typedef bf16_t v16bf __attribute__((ext_vector_type(16)));
typedef float  v8f   __attribute__((ext_vector_type(8)));

union Frag16 {
    v16bf v;
    unsigned int u[8];
    uint4 q[2];
};

__device__ __forceinline__ unsigned short f2bf(float f) {
    unsigned u = __builtin_bit_cast(unsigned, f);
    unsigned r = u + 0x7FFFu + ((u >> 16) & 1u);   // round-to-nearest-even
    return (unsigned short)(r >> 16);
}

__device__ __forceinline__ int reflect_idx(int i, int n) {
    if (i < 0) i = -i;
    if (i >= n) i = 2 * n - 2 - i;
    return i;
}

// Stage one dword-column slice of a 32k x 64n weight chunk into the
// transposed LDS tile ([n][k-pair], row stride 20 dwords). Each thread moves
// 4 dwords: global (row, n0+sq*4 .. +3) -> LDS rows sq*4..sq*4+3 at col sp.
__device__ __forceinline__ void stage_chunk(const unsigned int* __restrict__ src,
                                            unsigned int* dst) {
#ifdef ASYNC_STAGE
#pragma unroll
    for (int j = 0; j < 4; ++j) {
        __builtin_amdgcn_global_load_async_to_lds_b32(
            (int*)(src + j), (int*)(dst + j * 20), 0, 0);
    }
#else
    uint4 wv = *(const uint4*)src;
    dst[0]  = wv.x;
    dst[20] = wv.y;
    dst[40] = wv.z;
    dst[60] = wv.w;
#endif
}

__device__ __forceinline__ void wait_stage() {
#ifdef ASYNC_STAGE
#if __has_builtin(__builtin_amdgcn_s_wait_asynccnt)
    __builtin_amdgcn_s_wait_asynccnt(0);
#else
    asm volatile("s_wait_asynccnt 0x0" ::: "memory");
#endif
#endif
}

// ---------------------------------------------------------------------------
// Weight repack: (cout, cin, kh, kw) f32 -> pair-interleaved bf16
//   k = ((kh*Ksz + kw)*Cpadin + c) ; element (k,n) at
//   wb2[((k>>1)*Npad + n)*2 + (k&1)] ; zero-padded in k (c>=Cin) and n.
// ---------------------------------------------------------------------------
__global__ void convert_weights(const float* __restrict__ w,
                                unsigned short* __restrict__ wb2,
                                int Cin, int Cout, int Npad, int Ksz,
                                int Cpadin, int Kdim) {
    int idx = blockIdx.x * blockDim.x + threadIdx.x;
    int total = Kdim * Npad;
    if (idx >= total) return;
    int k = idx / Npad;
    int n = idx - k * Npad;
    int pos = k / Cpadin;
    int c = k - pos * Cpadin;
    int kh = pos / Ksz;
    int kw = pos - kh * Ksz;
    unsigned short v = 0;
    if (c < Cin && n < Cout)
        v = f2bf(w[(((size_t)n * Cin + c) * Ksz + kh) * Ksz + kw]);
    wb2[((size_t)(k >> 1) * Npad + n) * 2 + (k & 1)] = v;
}

// ---------------------------------------------------------------------------
// Input repack: f32 NCHW (B,3,H,W) -> bf16 NHWC (B,H,W,32) zero-padded.
// ---------------------------------------------------------------------------
__global__ void repack_input(const float* __restrict__ in,
                             unsigned short* __restrict__ out, int HW) {
    int idx = blockIdx.x * blockDim.x + threadIdx.x;
    int total = 2 * HW;
    if (idx >= total) return;
    int b = idx / HW;
    int pix = idx - b * HW;
    unsigned short* o = out + (size_t)idx * 32;
#pragma unroll
    for (int c = 0; c < 32; ++c)
        o[c] = (c < 3) ? f2bf(in[((size_t)b * 3 + c) * HW + pix]) : (unsigned short)0;
}

// ---------------------------------------------------------------------------
// Implicit-GEMM conv, WMMA bf16, double-buffered async weight staging.
//   input : bf16 NHWC [B][IH][IW][Cpadin]   (Cpadin % 32 == 0)
//   weight: packed bf16, K-order (kh,kw,c); pair-row q*16+p is chunk q
//   output: f32 NCHW raw (pre-norm), bias added
// Block = 256 thr = 8 waves. Wave: 32 pixels x 64 couts (8 accumulators).
// Pipeline per chunk: wait(asynccnt) -> barrier -> issue async stage of
// chunk q+1 into LDS buffer (q+1)&1 -> A loads + 8 WMMAs on buffer q&1.
// ---------------------------------------------------------------------------
__global__ __launch_bounds__(256) void conv_wmma(
    const unsigned short* __restrict__ act, const unsigned short* __restrict__ wb2,
    const float* __restrict__ bias, float* __restrict__ out,
    int Batch, int Cpadin, int Cout, int Npad, int Ksz, int stride, int pad,
    int IH, int IW, int OH, int OW, int upsample) {
    __shared__ unsigned int sBu[2][64 * 20];   // two transposed weight tiles

    const int tid    = threadIdx.x;
    const int wave   = tid >> 5;
    const int lane   = tid & 31;
    const int lane15 = lane & 15;
    const int hi     = lane >> 4;

    const int HWo  = OH * OW;
    const int Mtot = Batch * HWo;
    const int n0   = blockIdx.y * 64;
    const int m0   = (blockIdx.x * 8 + wave) * 32;

    // two pixel rows per lane (two M-tiles)
    int oy[2], ox[2], bb[2];
#pragma unroll
    for (int mt = 0; mt < 2; ++mt) {
        int m = m0 + mt * 16 + lane15;
        if (m >= Mtot) m = Mtot - 1;
        bb[mt] = m / HWo;
        int rem = m - bb[mt] * HWo;
        oy[mt] = rem / OW;
        ox[mt] = rem - oy[mt] * OW;
    }

    // staging coordinates
    const int sp = tid >> 4;   // k-pair row 0..15
    const int sq = tid & 15;   // n-quad 0..15
    const unsigned int* wb32 = (const unsigned int*)wb2;  // dword (p,n) view
    const int stageN = n0 + sq * 4;
    const int nchunks = Ksz * Ksz * (Cpadin >> 5);

    v8f zero = {0.f, 0.f, 0.f, 0.f, 0.f, 0.f, 0.f, 0.f};
    v8f acc[2][4] = {{zero, zero, zero, zero}, {zero, zero, zero, zero}};

    const int laneCh = hi * 8;

    // prologue: stage chunk 0 into buffer 0
    stage_chunk(wb32 + (size_t)sp * Npad + stageN, &sBu[0][(sq * 4) * 20 + sp]);

    int q = 0;
    for (int kh = 0; kh < Ksz; ++kh) {
        int iy[2];
#pragma unroll
        for (int mt = 0; mt < 2; ++mt) {
            if (upsample)
                iy[mt] = reflect_idx(oy[mt] + kh - pad, 2 * IH) >> 1;
            else
                iy[mt] = reflect_idx(oy[mt] * stride + kh - pad, IH);
        }
        for (int kw = 0; kw < Ksz; ++kw) {
            const unsigned short* abase[2];
#pragma unroll
            for (int mt = 0; mt < 2; ++mt) {
                int ix = upsample
                             ? (reflect_idx(ox[mt] + kw - pad, 2 * IW) >> 1)
                             : reflect_idx(ox[mt] * stride + kw - pad, IW);
                abase[mt] = act +
                            (((size_t)bb[mt] * IH + iy[mt]) * IW + ix) * Cpadin +
                            laneCh;
            }

            for (int cb = 0; cb < Cpadin; cb += 32) {
                wait_stage();        // chunk q async done (this wave)
                __syncthreads();     // all waves staged q / done reading q-1
                if (q + 1 < nchunks)
                    stage_chunk(
                        wb32 + (size_t)((q + 1) * 16 + sp) * Npad + stageN,
                        &sBu[(q + 1) & 1][(sq * 4) * 20 + sp]);

                // ---- A fragments: contiguous channels, 2 x b128 each ----
                Frag16 a[2];
#pragma unroll
                for (int mt = 0; mt < 2; ++mt) {
                    a[mt].q[0] = *(const uint4*)(abase[mt] + cb);
                    a[mt].q[1] = *(const uint4*)(abase[mt] + cb + 16);
                }

                // ---- 4 cout tiles from LDS buffer q&1 ----
                const unsigned int* sbuf = sBu[q & 1];
#pragma unroll
                for (int t = 0; t < 4; ++t) {
                    Frag16 b;
                    const unsigned int* bp = &sbuf[(t * 16 + lane15) * 20 + hi * 4];
                    b.q[0] = *(const uint4*)(bp);
                    b.q[1] = *(const uint4*)(bp + 8);
#pragma unroll
                    for (int mt = 0; mt < 2; ++mt)
                        acc[mt][t] = __builtin_amdgcn_wmma_f32_16x16x32_bf16(
                            false, a[mt].v, false, b.v, (short)0, acc[mt][t],
                            false, false);
                }
                ++q;
            }
        }
    }

    // ---- store D + bias (f32 NCHW, pre-norm) ----
    if (m0 < Mtot) {
#pragma unroll
        for (int t = 0; t < 4; ++t) {
            int col = n0 + t * 16 + lane15;
            if (col < Cout) {
                float bv = bias[col];
#pragma unroll
                for (int mt = 0; mt < 2; ++mt) {
#pragma unroll
                    for (int r = 0; r < 8; ++r) {
                        int m = m0 + mt * 16 + r + 8 * hi;
                        if (m < Mtot) {
                            int b = m / HWo;
                            int rem = m - b * HWo;
                            int oyS = rem / OW;
                            int oxS = rem - oyS * OW;
                            out[(((size_t)b * Cout + col) * OH + oyS) * OW + oxS] =
                                acc[mt][t][r] + bv;
                        }
                    }
                }
            }
        }
    }
}

// ---------------------------------------------------------------------------
// InstanceNorm stats: one block per (b, c)
// ---------------------------------------------------------------------------
__global__ __launch_bounds__(256) void instnorm_stats(
    const float* __restrict__ x, float* __restrict__ mean,
    float* __restrict__ rstd, int HW) {
    __shared__ float s1[256];
    __shared__ float s2[256];
    int bc = blockIdx.x;
    const float* p = x + (size_t)bc * HW;
    float sum = 0.f, sq = 0.f;
    for (int i = threadIdx.x; i < HW; i += 256) {
        float v = p[i];
        sum += v;
        sq += v * v;
    }
    s1[threadIdx.x] = sum;
    s2[threadIdx.x] = sq;
    __syncthreads();
    for (int off = 128; off > 0; off >>= 1) {
        if (threadIdx.x < off) {
            s1[threadIdx.x] += s1[threadIdx.x + off];
            s2[threadIdx.x] += s2[threadIdx.x + off];
        }
        __syncthreads();
    }
    if (threadIdx.x == 0) {
        float m = s1[0] / (float)HW;
        float var = s2[0] / (float)HW - m * m;
        mean[bc] = m;
        rstd[bc] = rsqrtf(var + 1e-5f);
    }
}

// ---------------------------------------------------------------------------
// Normalize + ReLU + NCHW->NHWC transpose through LDS (both sides coalesced).
// Grid: (HW/64, (B*Cout)/64), 256 threads. Cout % 64 == 0 here.
// ---------------------------------------------------------------------------
__global__ __launch_bounds__(256) void norm_act_transpose(
    const float* __restrict__ raw, const float* __restrict__ mean,
    const float* __restrict__ rstd, unsigned short* __restrict__ nhwc,
    int HW, int Cout) {
    __shared__ unsigned short tile[64][66];
    int pix0 = blockIdx.x * 64;
    int cb   = blockIdx.y * 64;          // channel block within B*Cout
    int b    = cb / Cout;
    int c0   = cb - b * Cout;
    int tx = threadIdx.x & 63;
    int ty = threadIdx.x >> 6;           // 0..3
#pragma unroll
    for (int i = 0; i < 16; ++i) {
        int cc = ty + i * 4;
        int bc = cb + cc;
        float v = raw[(size_t)bc * HW + pix0 + tx];
        v = (v - mean[bc]) * rstd[bc];
        tile[cc][tx] = f2bf(fmaxf(v, 0.f));
    }
    __syncthreads();
#pragma unroll
    for (int i = 0; i < 16; ++i) {
        int pp = ty + i * 4;
        nhwc[((size_t)b * HW + pix0 + pp) * Cout + c0 + tx] = tile[tx][pp];
    }
}

__global__ void tanh_inplace(float* __restrict__ x, int total) {
    int idx = blockIdx.x * blockDim.x + threadIdx.x;
    if (idx < total) x[idx] = tanhf(x[idx]);
}

// ---------------------------------------------------------------------------
// Instance-wise mean: LDS-reduced scatter, then broadcast (B=2, C=3, 16 segs)
// ---------------------------------------------------------------------------
__global__ void zero_f32(float* p, int n) {
    int i = blockIdx.x * blockDim.x + threadIdx.x;
    if (i < n) p[i] = 0.f;
}

__global__ __launch_bounds__(256) void inst_accum(
    const float* __restrict__ feats, const int* __restrict__ inst,
    float* __restrict__ sums, float* __restrict__ cnts, int HW) {
    __shared__ float ls[64];   // [seg][c0..2, cnt]
    int t = threadIdx.x;
    if (t < 64) ls[t] = 0.f;
    __syncthreads();
    int idx = blockIdx.x * 256 + t;       // over 2*HW; HW % 256 == 0
    int b = idx / HW;
    int pix = idx - b * HW;
    int seg = inst[idx];
    seg = min(max(seg, 0), 15);
    atomicAdd(&ls[seg * 4 + 3], 1.0f);
#pragma unroll
    for (int c = 0; c < 3; ++c)
        atomicAdd(&ls[seg * 4 + c], feats[(size_t)(b * 3 + c) * HW + pix]);
    __syncthreads();
    if (t < 64) {
        float v = ls[t];
        if (v != 0.f) {
            int s = t >> 2, c = t & 3;
            if (c < 3)
                atomicAdd(&sums[(b * 16 + s) * 3 + c], v);
            else
                atomicAdd(&cnts[b * 16 + s], v);
        }
    }
}

__global__ void inst_broadcast(const int* __restrict__ inst,
                               const float* __restrict__ sums,
                               const float* __restrict__ cnts,
                               float* __restrict__ out, int HW) {
    int idx = blockIdx.x * blockDim.x + threadIdx.x;
    int total = 2 * 3 * HW;
    if (idx >= total) return;
    int b = idx / (3 * HW);
    int r = idx - b * 3 * HW;
    int c = r / HW;
    int pix = r - c * HW;
    int seg = inst[(size_t)b * HW + pix];
    seg = min(max(seg, 0), 15);
    float cnt = fmaxf(cnts[b * 16 + seg], 1.f);
    out[idx] = sums[(b * 16 + seg) * 3 + c] / cnt;
}

// ---------------------------------------------------------------------------
// Host orchestration
// ---------------------------------------------------------------------------
struct LayerCfg {
    int cin, cout, k, stride, pad, nrm, up, tanh_;
};
static const LayerCfg LCFG[10] = {
    {3, 64, 7, 1, 3, 1, 0, 0},     {64, 128, 3, 2, 1, 1, 0, 0},
    {128, 256, 3, 2, 1, 1, 0, 0},  {256, 512, 3, 2, 1, 1, 0, 0},
    {512, 1024, 3, 2, 1, 1, 0, 0}, {1024, 512, 3, 1, 1, 1, 1, 0},
    {512, 256, 3, 1, 1, 1, 1, 0},  {256, 128, 3, 1, 1, 1, 1, 0},
    {128, 64, 3, 1, 1, 1, 1, 0},   {64, 3, 7, 1, 3, 0, 0, 1}};

extern "C" void kernel_launch(void* const* d_in, const int* in_sizes, int n_in,
                              void* d_out, int out_size, void* d_ws,
                              size_t ws_size, hipStream_t stream) {
    (void)in_sizes; (void)n_in; (void)ws_size;
    const int Batch = 2;
    char* ws = (char*)d_ws;

    const size_t RAW_BYTES  = (size_t)Batch * 64 * 512 * 512 * sizeof(float); // 128MB
    const size_t NHWC_BYTES = (size_t)Batch * 512 * 512 * 64 * 2;             // 64MB

    float* rawBuf = (float*)ws;
    unsigned short* nhwc[2] = {(unsigned short*)(ws + RAW_BYTES),
                               (unsigned short*)(ws + RAW_BYTES + NHWC_BYTES)};
    char* wbase = ws + RAW_BYTES + 2 * NHWC_BYTES;

    size_t woff[10];
    size_t wtot = 0;
    int cpin[10], Kdim[10], Npad[10];
    for (int i = 0; i < 10; ++i) {
        cpin[i] = (i == 0) ? 32 : LCFG[i].cin;             // % 32 == 0
        Kdim[i] = LCFG[i].k * LCFG[i].k * cpin[i];         // % 32 == 0
        Npad[i] = ((LCFG[i].cout + 63) / 64) * 64;
        woff[i] = wtot;
        wtot += (size_t)Kdim[i] * Npad[i] * sizeof(unsigned short);
    }
    float* meanBuf = (float*)(wbase + wtot);
    float* rstdBuf = meanBuf + 2048;
    float* sums = rstdBuf + 2048;      // 2*16*3
    float* cnts = sums + 2 * 16 * 3;   // 2*16

    // 1) repack all weights
    for (int i = 0; i < 10; ++i) {
        const float* w = (const float*)d_in[1 + 2 * i];
        unsigned short* wb2 = (unsigned short*)(wbase + woff[i]);
        int total = Kdim[i] * Npad[i];
        convert_weights<<<(total + 255) / 256, 256, 0, stream>>>(
            w, wb2, LCFG[i].cin, LCFG[i].cout, Npad[i], LCFG[i].k, cpin[i],
            Kdim[i]);
    }

    // 2) input image -> bf16 NHWC (channels padded to 32)
    repack_input<<<(2 * 512 * 512 + 255) / 256, 256, 0, stream>>>(
        (const float*)d_in[0], nhwc[0], 512 * 512);

    // 3) backbone
    int curH = 512, curW = 512;
    int pp = 0;  // nhwc[pp] holds current layer input
    for (int i = 0; i < 10; ++i) {
        const LayerCfg& c = LCFG[i];
        int effH = c.up ? curH * 2 : curH;
        int effW = c.up ? curW * 2 : curW;
        int OH = effH / c.stride;
        int OW = effW / c.stride;

        int Mtot = Batch * OH * OW;           // multiple of 32
        int mtiles32 = Mtot / 32;
        dim3 grid((mtiles32 + 7) / 8, Npad[i] / 64);
        conv_wmma<<<grid, 256, 0, stream>>>(
            nhwc[pp], (const unsigned short*)(wbase + woff[i]),
            (const float*)d_in[2 + 2 * i], rawBuf, Batch, cpin[i], c.cout,
            Npad[i], c.k, c.stride, c.pad, curH, curW, OH, OW, c.up);

        int HW = OH * OW;
        if (i < 9) {
            instnorm_stats<<<Batch * c.cout, 256, 0, stream>>>(rawBuf, meanBuf,
                                                               rstdBuf, HW);
            dim3 ngrid(HW / 64, (Batch * c.cout) / 64);
            norm_act_transpose<<<ngrid, 256, 0, stream>>>(
                rawBuf, meanBuf, rstdBuf, nhwc[pp ^ 1], HW, c.cout);
        } else {
            int total = Batch * c.cout * HW;
            tanh_inplace<<<(total + 255) / 256, 256, 0, stream>>>(rawBuf, total);
        }

        curH = OH;
        curW = OW;
        pp ^= 1;
    }

    // 4) instance-wise mean broadcast (rawBuf holds tanh'd (2,3,512,512))
    const int* inst = (const int*)d_in[21];
    int HW = 512 * 512;
    zero_f32<<<1, 256, 0, stream>>>(sums, 2 * 16 * 3 + 2 * 16);
    inst_accum<<<(2 * HW) / 256, 256, 0, stream>>>(rawBuf, inst, sums, cnts, HW);
    inst_broadcast<<<(out_size + 255) / 256, 256, 0, stream>>>(
        inst, sums, cnts, (float*)d_out, HW);
}
